// ManyNetworkModel_24111946399932
// MI455X (gfx1250) — matrix operand order repaired and checked by reading, not compiled, not run on hardware
//
#include <hip/hip_runtime.h>

// MI455X / gfx1250: wave32, WMMA 16x16x32 f16->f32
// Computes C^T = W1[f]^T (H x I) * fa^T (I x rows): M = hidden, N = batch rows.
// This makes the H-reduction of the second layer in-lane (one shfl_xor(16) total).
typedef __attribute__((ext_vector_type(16))) _Float16 v16h;
typedef __attribute__((ext_vector_type(8)))  _Float16 v8h;
typedef __attribute__((ext_vector_type(8)))  float    v8f;

#define B_TOTAL  131072
#define NF       32          // number of submodels / output features
#define NA       8           // actions
#define NI       40          // F + A (real K)
#define KPAD     64          // padded K (two 16x16x32 WMMA steps)
#define NH       64          // hidden
#define ROWS     128         // batch rows per block
#define THREADS  256         // 8 wave32
#define RES_STRIDE 129       // bank-conflict-free [f][row] result layout

__device__ __forceinline__ v16h cat8h(v8h lo, v8h hi) {
    return __builtin_shufflevector(lo, hi, 0,1,2,3,4,5,6,7,8,9,10,11,12,13,14,15);
}

// Stage W1[f]^T (f16, [h][i_pad64]) into LDS: 16 halves per thread, two b128 stores.
__device__ __forceinline__ void fill_w(_Float16* __restrict__ dst,
                                       const float* __restrict__ W1f, int tid) {
    const int n  = tid >> 2;         // hidden index 0..63
    const int i0 = (tid & 3) << 4;   // input chunk base: 0,16,32,48
    __attribute__((aligned(16))) _Float16 tmp[16];
#pragma unroll
    for (int k = 0; k < 16; ++k) {
        const int i = i0 + k;
        tmp[k] = (i < NI) ? (_Float16)W1f[i * NH + n] : (_Float16)0.0f;
    }
    *(v8h*)&dst[n * KPAD + i0]     = *(const v8h*)&tmp[0];
    *(v8h*)&dst[n * KPAD + i0 + 8] = *(const v8h*)&tmp[8];
}

__global__ __launch_bounds__(THREADS)
void many_net_wmma_kernel(const float* __restrict__ f_t,
                          const float* __restrict__ a_t,
                          const float* __restrict__ W1,   // [32][40][64]
                          const float* __restrict__ b1,   // [32][64]
                          const float* __restrict__ W2,   // [32][64]
                          const float* __restrict__ b2,   // [32]
                          float* __restrict__ out)        // [B][32]
{
    __shared__ __attribute__((aligned(16))) _Float16 s_fa[ROWS * KPAD];      // 16 KB
    __shared__ __attribute__((aligned(16))) _Float16 s_w[2 * NH * KPAD];     // 16 KB (dbl buf)
    __shared__ __attribute__((aligned(16))) float    s_b1[NF * NH];          //  8 KB
    __shared__ __attribute__((aligned(16))) float    s_w2[NF * NH];          //  8 KB
    __shared__ __attribute__((aligned(16))) float    s_res[NF * RES_STRIDE]; // ~16 KB

    const int rowbase = blockIdx.x * ROWS;
    const int tid  = threadIdx.x;
    const int lane = tid & 31;
    const int wave = tid >> 5;
    const int nl      = lane & 15;        // N (batch row) / M (h) selector within tile
    const int halfsel = (lane >> 4) & 1;  // lane-half selector

    // ---- Stage fa = [f_t | a_t | 0-pad] as f16 into LDS (coalesced global reads) ----
    for (int idx = tid; idx < ROWS * KPAD; idx += THREADS) {
        const int r = idx >> 6;           // local row
        const int k = idx & 63;           // padded input index
        const int row = rowbase + r;
        float v = 0.0f;
        if (k < NF)       v = f_t[row * NF + k];
        else if (k < NI)  v = a_t[row * NA + (k - NF)];
        s_fa[idx] = (_Float16)v;
    }
    // ---- Stage b1 / W2 (f32) into LDS ----
    for (int idx = tid; idx < NF * NH; idx += THREADS) {
        s_b1[idx] = b1[idx];
        s_w2[idx] = W2[idx];
    }
    // ---- Prime W1 double buffer with f=0 ----
    fill_w(&s_w[0], W1, tid);
    __syncthreads();

    // ---- f-invariant B fragments: fa^T, B(K=i, N=row) ----
    // B 32x16 k order used here: lanes0-15 K=0..15, lanes16-31 K=16..31 (contiguous, same as A)
    const int brow = wave * 16 + nl;
    v16h bfrag[2];
#pragma unroll
    for (int s = 0; s < 2; ++s) {
        const int k0 = s * 32 + halfsel * 16;
        v8h lo = *(const v8h*)&s_fa[brow * KPAD + k0];
        v8h hi = *(const v8h*)&s_fa[brow * KPAD + k0 + 8];
        bfrag[s] = cat8h(lo, hi);
    }

    for (int f = 0; f < NF; ++f) {
        // ---- Prefetch W1[f+1]^T into the other buffer (overlaps with WMMA below) ----
        if (f + 1 < NF)
            fill_w(&s_w[((f + 1) & 1) * (NH * KPAD)], W1 + (f + 1) * (NI * NH), tid);

        const _Float16* wbuf = &s_w[(f & 1) * (NH * KPAD)];

        // ---- C^T (64h x 16rows) via 8x v_wmma_f32_16x16x32_f16 ----
        // A = W1^T tile: A(M=h, K=i); 16-bit A layout: lanes0-15 K=0..15, lanes16-31 K=16..31 (contiguous, matches B)
        v8f c[4] = {};
#pragma unroll
        for (int s = 0; s < 2; ++s) {
#pragma unroll
            for (int t = 0; t < 4; ++t) {
                const int h  = t * 16 + nl;
                const int c0 = s * 32 + halfsel * 16;
                v8h lo = *(const v8h*)&wbuf[h * KPAD + c0];
                v8h hi = *(const v8h*)&wbuf[h * KPAD + c0 + 8];
                c[t] = __builtin_amdgcn_wmma_f32_16x16x32_f16(
                        /*neg_a=*/false, cat8h(lo, hi),
                        /*neg_b=*/false, bfrag[s],
                        /*c_mod=*/(short)0, c[t],
                        /*reuse_a=*/false, /*reuse_b=*/false);
            }
        }

        // ---- Epilogue: relu(c + b1) . W2, reduction over H is IN-LANE ----
        // c[t][r] holds h = t*16 + halfsel*8 + r for batch row (wave*16 + nl).
        float acc = 0.0f;
#pragma unroll
        for (int t = 0; t < 4; ++t) {
            const int hb = f * NH + t * 16 + halfsel * 8;
            const float4 b1a = *(const float4*)&s_b1[hb];
            const float4 b1b = *(const float4*)&s_b1[hb + 4];
            const float4 w2a = *(const float4*)&s_w2[hb];
            const float4 w2b = *(const float4*)&s_w2[hb + 4];
            const float vb1[8] = {b1a.x, b1a.y, b1a.z, b1a.w, b1b.x, b1b.y, b1b.z, b1b.w};
            const float vw2[8] = {w2a.x, w2a.y, w2a.z, w2a.w, w2b.x, w2b.y, w2b.z, w2b.w};
#pragma unroll
            for (int r = 0; r < 8; ++r) {
                float hh = c[t][r] + vb1[r];
                hh = hh > 0.0f ? hh : 0.0f;
                acc = fmaf(hh, vw2[r], acc);
            }
        }
        // merge the two lane-halves (other 32 h-values) — single cross-lane op
        acc += __shfl_xor(acc, 16, 32);
        if (lane < 16)
            s_res[f * RES_STRIDE + wave * 16 + lane] = acc;

        __syncthreads();  // fill(f+1) done; compute(f) done before buffer reuse
    }

    // ---- Add b2 + skip (f_t) and stream out coalesced ----
    for (int idx = tid; idx < ROWS * NF; idx += THREADS) {
        const int rl = idx >> 5;
        const int f  = idx & 31;
        const int row = rowbase + rl;
        out[row * NF + f] = s_res[f * RES_STRIDE + rl] + b2[f] + f_t[row * NF + f];
    }
}

extern "C" void kernel_launch(void* const* d_in, const int* in_sizes, int n_in,
                              void* d_out, int out_size, void* d_ws, size_t ws_size,
                              hipStream_t stream) {
    (void)in_sizes; (void)n_in; (void)out_size; (void)d_ws; (void)ws_size;
    const float* f_t = (const float*)d_in[0];
    const float* a_t = (const float*)d_in[1];
    const float* W1  = (const float*)d_in[2];
    const float* b1  = (const float*)d_in[3];
    const float* W2  = (const float*)d_in[4];
    const float* b2  = (const float*)d_in[5];
    float* out = (float*)d_out;

    dim3 grid(B_TOTAL / ROWS);   // 1024
    dim3 block(THREADS);         // 256 = 8 wave32
    many_net_wmma_kernel<<<grid, block, 0, stream>>>(f_t, a_t, W1, b1, W2, b2, out);
}
